// Decoder_60748017434951
// MI455X (gfx1250) — matrix operand order repaired and checked by reading, not compiled, not run on hardware
//
#include <hip/hip_runtime.h>
#include <hip/hip_bf16.h>

typedef __attribute__((ext_vector_type(16))) _Float16 v16h;
typedef __attribute__((ext_vector_type(8)))  _Float16 v8h;
typedef __attribute__((ext_vector_type(8)))  float    v8f;

// ---------------------------------------------------------------------------
// Conversion kernels
// ---------------------------------------------------------------------------
__global__ __launch_bounds__(256) void k_cvt16(const float* __restrict__ s,
                                               _Float16* __restrict__ d, long long n) {
  long long i = (long long)blockIdx.x * 256 + threadIdx.x;
  if (i < n) d[i] = (_Float16)s[i];
}

// W [K,N] f32 row-major  ->  Wt [N,K] f16  (WMMA B-fragment friendly layout)
__global__ __launch_bounds__(256) void k_cvtT(const float* __restrict__ W,
                                              _Float16* __restrict__ Wt,
                                              int K, int N) {
  long long i = (long long)blockIdx.x * 256 + threadIdx.x;
  if (i >= (long long)K * N) return;
  int n = (int)(i / K);
  int k = (int)(i % K);
  Wt[i] = (_Float16)W[(long long)k * N + n];
}

// ---------------------------------------------------------------------------
// Universal WMMA GEMM: C[M,N] = act(scale * A[M,K] @ Bt[N,K]^T + bias)
//   A row-major f16, Bt "transposed" f16 [N,K].
//   One wave computes a 64x64 tile via 4x4 grid of v_wmma_f32_16x16x32_f16.
//   Batched over blockIdx.z with linear A/B strides; C offset =
//   (z/hdiv)*c0 + (z%hdiv)*c1 to express per-(b,h) head scatter.
//   mode 0: idx = offC + m*ldc + n
//   mode 1: idx = offC + (n>>6)*hstr + m*64 + (n&63)        (Q/K  [B,H,S,dh])
//   mode 2: idx = offC + (n>>6)*hstr + (n&63)*sdim + m      (V    [B,H,dh,S])
// ---------------------------------------------------------------------------
__global__ __launch_bounds__(128) void k_wmma_gemm(
    const _Float16* __restrict__ A, long long sAz,
    const _Float16* __restrict__ Bt, long long sBz,
    const float* __restrict__ bias, float scale, int relu,
    float* __restrict__ Cf, _Float16* __restrict__ Ch,
    int mode, long long c0, long long c1, int hdiv,
    int ldc, long long hstr, int sdim,
    int M, int N, int Kd) {
  const int wave = threadIdx.x >> 5;
  const int lane = threadIdx.x & 31;
  const int tilesN = N >> 6;
  const int tilesM = M >> 6;
  const int tileId = blockIdx.x * 4 + wave;
  if (tileId >= tilesM * tilesN) return;
  const int tm = (tileId / tilesN) * 64;
  const int tn = (tileId % tilesN) * 64;
  const int z = blockIdx.z;

  const _Float16* Az = A + (long long)z * sAz;
  const _Float16* Bz = Bt + (long long)z * sBz;

  const int row = lane & 15;      // A: row-in-tile / B,C,D: col-in-tile
  const int hi  = lane >> 4;      // lane-half
  const int kA  = hi * 8;         // A K-offset within 32-block
  const int kB  = hi * 8;         // B K-offset within 32-block

  const v8f vzero = {0.f, 0.f, 0.f, 0.f, 0.f, 0.f, 0.f, 0.f};
  v8f acc[4][4];
#pragma unroll
  for (int i = 0; i < 4; ++i)
#pragma unroll
    for (int j = 0; j < 4; ++j) acc[i][j] = vzero;

  for (int k0 = 0; k0 < Kd; k0 += 32) {
    v16h a[4], b[4];
#pragma unroll
    for (int i = 0; i < 4; ++i) {
      const _Float16* pa = Az + (long long)(tm + i * 16 + row) * Kd + k0 + kA;
      v8h lo = *(const v8h*)pa;
      v8h hh = *(const v8h*)(pa + 16);
      a[i] = __builtin_shufflevector(lo, hh, 0, 1, 2, 3, 4, 5, 6, 7,
                                     8, 9, 10, 11, 12, 13, 14, 15);
      const _Float16* pb = Bz + (long long)(tn + i * 16 + row) * Kd + k0 + kB;
      v8h blo = *(const v8h*)pb;
      v8h bhi = *(const v8h*)(pb + 16);
      b[i] = __builtin_shufflevector(blo, bhi, 0, 1, 2, 3, 4, 5, 6, 7,
                                     8, 9, 10, 11, 12, 13, 14, 15);
      __builtin_prefetch(pa + 32, 0, 1);   // global_prefetch next K-block
      __builtin_prefetch(pb + 32, 0, 1);
    }
#pragma unroll
    for (int i = 0; i < 4; ++i)
#pragma unroll
      for (int j = 0; j < 4; ++j)
        acc[i][j] = __builtin_amdgcn_wmma_f32_16x16x32_f16(
            false, a[i], false, b[j], (short)0, acc[i][j], false, false);
  }

  const long long offC = (long long)(z / hdiv) * c0 + (long long)(z % hdiv) * c1;
#pragma unroll
  for (int i = 0; i < 4; ++i) {
#pragma unroll
    for (int j = 0; j < 4; ++j) {
      const int n = tn + j * 16 + row;
      const float bv = bias ? bias[n] : 0.f;
#pragma unroll
      for (int r = 0; r < 8; ++r) {
        const int m = tm + i * 16 + r + hi * 8;
        float v = acc[i][j][r] * scale + bv;
        if (relu) v = fmaxf(v, 0.f);
        long long idx;
        if (mode == 0)      idx = offC + (long long)m * ldc + n;
        else if (mode == 1) idx = offC + (long long)(n >> 6) * hstr + (long long)m * 64 + (n & 63);
        else                idx = offC + (long long)(n >> 6) * hstr + (long long)(n & 63) * sdim + m;
        if (Cf) Cf[idx] = v;
        if (Ch) Ch[idx] = (_Float16)v;
      }
    }
  }
}

// ---------------------------------------------------------------------------
// Block reductions (256 threads)
// ---------------------------------------------------------------------------
__device__ __forceinline__ float blockSum(float v, float* red) {
  int t = threadIdx.x;
  red[t] = v; __syncthreads();
  for (int s = 128; s > 0; s >>= 1) { if (t < s) red[t] += red[t + s]; __syncthreads(); }
  float r = red[0]; __syncthreads(); return r;
}
__device__ __forceinline__ float blockMax(float v, float* red) {
  int t = threadIdx.x;
  red[t] = v; __syncthreads();
  for (int s = 128; s > 0; s >>= 1) { if (t < s) red[t] = fmaxf(red[t], red[t + s]); __syncthreads(); }
  float r = red[0]; __syncthreads(); return r;
}

// y = LayerNorm(x + s); also writes f16 mirror. Row length fixed at 512.
__global__ __launch_bounds__(256) void k_add_ln(const float* __restrict__ x,
                                                const float* __restrict__ s,
                                                const float* __restrict__ g,
                                                const float* __restrict__ b,
                                                float* __restrict__ y,
                                                _Float16* __restrict__ y16) {
  __shared__ float red[256];
  long long row = blockIdx.x;
  int t = threadIdx.x;
  const float* xr = x + row * 512;
  const float* sr = s + row * 512;
  float v0 = xr[t] + sr[t];
  float v1 = xr[t + 256] + sr[t + 256];
  float mean = blockSum(v0 + v1, red) * (1.f / 512.f);
  float d0 = v0 - mean, d1 = v1 - mean;
  float var = blockSum(d0 * d0 + d1 * d1, red) * (1.f / 512.f);
  float inv = rsqrtf(var + 1e-5f);
  float o0 = d0 * inv * g[t] + b[t];
  float o1 = d1 * inv * g[t + 256] + b[t + 256];
  y[row * 512 + t] = o0;         y[row * 512 + t + 256] = o1;
  y16[row * 512 + t] = (_Float16)o0; y16[row * 512 + t + 256] = (_Float16)o1;
}

// softmax over rows of 512 (f16 in -> f16 out, optional f32 stream to d_out)
__global__ __launch_bounds__(256) void k_softmax(const _Float16* __restrict__ S,
                                                 _Float16* __restrict__ A16,
                                                 float* __restrict__ Af) {
  __shared__ float red[256];
  long long row = blockIdx.x;
  int t = threadIdx.x;
  const _Float16* sr = S + row * 512;
  float v0 = (float)sr[t], v1 = (float)sr[t + 256];
  float m = blockMax(fmaxf(v0, v1), red);
  float e0 = __expf(v0 - m), e1 = __expf(v1 - m);
  float inv = 1.f / blockSum(e0 + e1, red);
  e0 *= inv; e1 *= inv;
  A16[row * 512 + t] = (_Float16)e0;
  A16[row * 512 + t + 256] = (_Float16)e1;
  if (Af) { Af[row * 512 + t] = e0; Af[row * 512 + t + 256] = e1; }
}

// L2 norms of rows of 512
__global__ __launch_bounds__(256) void k_norm(const float* __restrict__ x,
                                              float* __restrict__ norms) {
  __shared__ float red[256];
  long long row = blockIdx.x;
  int t = threadIdx.x;
  const float* xr = x + row * 512;
  float v0 = xr[t], v1 = xr[t + 256];
  float s = blockSum(v0 * v0 + v1 * v1, red);
  if (t == 0) norms[row] = sqrtf(s);
}

// pooled[b,:] = sum_s softmax_s(norm[b,s]) * x[b,s,:]
__global__ __launch_bounds__(256) void k_pool(const float* __restrict__ x,
                                              const float* __restrict__ norms,
                                              float* __restrict__ pooled) {
  __shared__ float w[512];
  __shared__ float red[256];
  int b = blockIdx.x, t = threadIdx.x;
  float n0 = norms[b * 512 + t], n1 = norms[b * 512 + t + 256];
  float m = blockMax(fmaxf(n0, n1), red);
  float e0 = __expf(n0 - m), e1 = __expf(n1 - m);
  float inv = 1.f / blockSum(e0 + e1, red);
  w[t] = e0 * inv; w[t + 256] = e1 * inv;
  __syncthreads();
  for (int d = t; d < 512; d += 256) {
    float acc = 0.f;
    const float* xb = x + (long long)b * 512 * 512 + d;
    for (int s2 = 0; s2 < 512; ++s2) acc += xb[(long long)s2 * 512] * w[s2];
    pooled[b * 512 + d] = acc;
  }
}

// tiny classifier head: 512 ->256 relu ->64 relu ->2
__global__ __launch_bounds__(256) void k_head(const float* __restrict__ pooled,
                                              const float* __restrict__ w1, const float* __restrict__ b1,
                                              const float* __restrict__ w2, const float* __restrict__ b2,
                                              const float* __restrict__ w3, const float* __restrict__ b3,
                                              float* __restrict__ label) {
  __shared__ float p[512], h1[256], h2[64];
  int b = blockIdx.x, t = threadIdx.x;
  for (int i = t; i < 512; i += 256) p[i] = pooled[b * 512 + i];
  __syncthreads();
  float a = b1[t];
  for (int k = 0; k < 512; ++k) a += p[k] * w1[k * 256 + t];
  h1[t] = fmaxf(a, 0.f);
  __syncthreads();
  if (t < 64) {
    float a2 = b2[t];
    for (int k = 0; k < 256; ++k) a2 += h1[k] * w2[k * 64 + t];
    h2[t] = fmaxf(a2, 0.f);
  }
  __syncthreads();
  if (t < 2) {
    float a3 = b3[t];
    for (int k = 0; k < 64; ++k) a3 += h2[k] * w3[k * 2 + t];
    label[b * 2 + t] = a3;
  }
}

// ---------------------------------------------------------------------------
extern "C" void kernel_launch(void* const* d_in, const int* in_sizes, int n_in,
                              void* d_out, int out_size, void* d_ws, size_t ws_size,
                              hipStream_t stream) {
  (void)in_sizes; (void)n_in; (void)out_size; (void)ws_size;
  constexpr int Bc = 16, ST = 512, SS = 512, LOC = 1024, D = 512, FF = 2048, L = 3, H = 8, DH = 64;
  constexpr long long BS = (long long)Bc * ST;      // 8192 rows

  const float* trg  = (const float*)d_in[0];
  const float* src  = (const float*)d_in[1];
  const float* ft_w = (const float*)d_in[2];
  const float* ft_b = (const float*)d_in[3];
  const float *sa_w[4], *sa_b[4], *ea_w[4], *ea_b[4];
  for (int i = 0; i < 4; ++i) {
    sa_w[i] = (const float*)d_in[4 + 2 * i];  sa_b[i] = (const float*)d_in[5 + 2 * i];
    ea_w[i] = (const float*)d_in[12 + 2 * i]; ea_b[i] = (const float*)d_in[13 + 2 * i];
  }
  const float* pf_w1 = (const float*)d_in[20]; const float* pf_b1 = (const float*)d_in[21];
  const float* pf_w2 = (const float*)d_in[22]; const float* pf_b2 = (const float*)d_in[23];
  const float* ln_g[3] = {(const float*)d_in[24], (const float*)d_in[26], (const float*)d_in[28]};
  const float* ln_b[3] = {(const float*)d_in[25], (const float*)d_in[27], (const float*)d_in[29]};
  const float* fc1_w = (const float*)d_in[30]; const float* fc1_b = (const float*)d_in[31];
  const float* fc2_w = (const float*)d_in[32]; const float* fc2_b = (const float*)d_in[33];
  const float* fc3_w = (const float*)d_in[34]; const float* fc3_b = (const float*)d_in[35];

  float* outPooled = (float*)d_out;
  float* outAttn   = outPooled + (long long)Bc * D;
  float* outLabel  = outAttn + (long long)Bc * H * ST * SS;

  // ---- workspace bump allocator ----
  char* wp = (char*)d_ws;
  auto alloc = [&](size_t bytes) -> void* {
    void* p = (void*)wp; wp += (bytes + 255) & ~(size_t)255; return p;
  };
  _Float16* trg16 = (_Float16*)alloc(BS * LOC * 2);
  _Float16* src16 = (_Float16*)alloc((long long)Bc * SS * D * 2);
  _Float16* ftT   = (_Float16*)alloc((long long)D * LOC * 2);        // [D, LOC]
  _Float16 *saT[4], *eaT[4];
  for (int i = 0; i < 4; ++i) saT[i] = (_Float16*)alloc((long long)L * D * D * 2);
  for (int i = 0; i < 4; ++i) eaT[i] = (_Float16*)alloc((long long)L * D * D * 2);
  _Float16* w1T = (_Float16*)alloc((long long)L * D * FF * 2);       // [FF, D] per layer
  _Float16* w2T = (_Float16*)alloc((long long)L * FF * D * 2);       // [D, FF] per layer
  _Float16* x16   = (_Float16*)alloc(BS * D * 2);
  _Float16* q16   = (_Float16*)alloc((long long)Bc * H * ST * DH * 2);
  _Float16* kk16  = (_Float16*)alloc((long long)Bc * H * SS * DH * 2);
  _Float16* v16   = (_Float16*)alloc((long long)Bc * H * DH * SS * 2);
  _Float16* ctx16 = (_Float16*)alloc(BS * D * 2);
  _Float16* h16   = (_Float16*)alloc(BS * FF * 2);
  _Float16* sc16  = (_Float16*)alloc((long long)Bc * H * ST * SS * 2);
  _Float16* at16  = (_Float16*)alloc((long long)Bc * H * ST * SS * 2);
  float* xf0   = (float*)alloc(BS * D * 4);
  float* xf1   = (float*)alloc(BS * D * 4);
  float* sf    = (float*)alloc(BS * D * 4);
  float* norms = (float*)alloc(BS * 4);

  auto cvt = [&](const float* s, _Float16* d, long long n) {
    k_cvt16<<<dim3((unsigned)((n + 255) / 256)), 256, 0, stream>>>(s, d, n);
  };
  auto cvtT = [&](const float* W, _Float16* Wt, int K, int N) {
    long long n = (long long)K * N;
    k_cvtT<<<dim3((unsigned)((n + 255) / 256)), 256, 0, stream>>>(W, Wt, K, N);
  };
  auto gemm = [&](const _Float16* A, long long sAz, const _Float16* Bt, long long sBz,
                  const float* bias, float scale, int relu,
                  float* Cf, _Float16* Ch, int mode,
                  long long c0, long long c1, int hdiv, int ldc, long long hstr, int sdim,
                  int M, int N, int Kd, int Z) {
    int tiles = (M / 64) * (N / 64);
    dim3 grid((tiles + 3) / 4, 1, Z);
    k_wmma_gemm<<<grid, 128, 0, stream>>>(A, sAz, Bt, sBz, bias, scale, relu, Cf, Ch,
                                          mode, c0, c1, hdiv, ldc, hstr, sdim, M, N, Kd);
  };

  // ---- one-time conversions ----
  cvt(trg, trg16, BS * LOC);
  cvt(src, src16, (long long)Bc * SS * D);
  cvtT(ft_w, ftT, LOC, D);
  for (int l = 0; l < L; ++l) {
    for (int i = 0; i < 4; ++i) {
      cvtT(sa_w[i] + (long long)l * D * D, saT[i] + (long long)l * D * D, D, D);
      cvtT(ea_w[i] + (long long)l * D * D, eaT[i] + (long long)l * D * D, D, D);
    }
    cvtT(pf_w1 + (long long)l * D * FF, w1T + (long long)l * D * FF, D, FF);
    cvtT(pf_w2 + (long long)l * FF * D, w2T + (long long)l * FF * D, FF, D);
  }

  // ---- feature projection: x = trg @ ft_w + ft_b ----
  gemm(trg16, 0, ftT, 0, ft_b, 1.f, 0, xf0, x16,
       0, 0, 0, 1, D, 0, 0, (int)BS, D, LOC, 1);

  float* cur = xf0;
  float* nxt = xf1;
  const long long hBlk = (long long)ST * DH;          // 32768: per-(b,h) block
  const long long bBlk = (long long)H * ST * DH;      // 262144: per-b block

  for (int l = 0; l < L; ++l) {
    const long long wOff = (long long)l * D * D;
    // ======== attention (pass 0: self on x16/x16, pass 1: cross on x16/src16)
    for (int pass = 0; pass < 2; ++pass) {
      const _Float16* kvIn = pass ? src16 : x16;
      _Float16* const* wT  = pass ? eaT : saT;
      const float* const* bb = pass ? ea_b : sa_b;
      // Q/K/V projections with per-head scatter
      gemm(x16, (long long)ST * D, wT[0] + wOff, 0, bb[0] + (long long)l * D, 1.f, 0,
           nullptr, q16, 1, bBlk, 0, 1, D, hBlk, 0, ST, D, D, Bc);
      gemm(kvIn, (long long)SS * D, wT[1] + wOff, 0, bb[1] + (long long)l * D, 1.f, 0,
           nullptr, kk16, 1, bBlk, 0, 1, D, hBlk, 0, SS, D, D, Bc);
      gemm(kvIn, (long long)SS * D, wT[2] + wOff, 0, bb[2] + (long long)l * D, 1.f, 0,
           nullptr, v16, 2, bBlk, 0, 1, D, hBlk, SS, SS, D, D, Bc);
      // scores = Q K^T / 8   (K in [S,dh] is already B-transposed layout)
      gemm(q16, hBlk, kk16, hBlk, nullptr, 0.125f, 0,
           nullptr, sc16, 0, (long long)ST * SS, 0, 1, SS, 0, 0, ST, SS, DH, Bc * H);
      // softmax; last-layer cross-attention also streams f32 attn to d_out
      float* af = (pass == 1 && l == L - 1) ? outAttn : nullptr;
      k_softmax<<<dim3((unsigned)(Bc * H * ST)), 256, 0, stream>>>(sc16, at16, af);
      // ctx = attn @ V  (V stored [dh,S] = B-transposed), scatter to [B,Sq,D]
      gemm(at16, (long long)ST * SS, v16, (long long)DH * SS, nullptr, 1.f, 0,
           nullptr, ctx16, 0, (long long)ST * D, DH, H, D, 0, 0, ST, DH, SS, Bc * H);
      // O projection -> sf
      gemm(ctx16, 0, wT[3] + wOff, 0, bb[3] + (long long)l * D, 1.f, 0,
           sf, nullptr, 0, 0, 0, 1, D, 0, 0, (int)BS, D, D, 1);
      // x = LN(x + s)
      k_add_ln<<<dim3((unsigned)BS), 256, 0, stream>>>(
          cur, sf, ln_g[pass] + (long long)l * D, ln_b[pass] + (long long)l * D, nxt, x16);
      float* tmp = cur; cur = nxt; nxt = tmp;
    }
    // ======== FFN
    gemm(x16, 0, w1T + (long long)l * D * FF, 0, pf_b1 + (long long)l * FF, 1.f, 1,
         nullptr, h16, 0, 0, 0, 1, FF, 0, 0, (int)BS, FF, D, 1);
    gemm(h16, 0, w2T + (long long)l * FF * D, 0, pf_b2 + (long long)l * D, 1.f, 0,
         sf, nullptr, 0, 0, 0, 1, D, 0, 0, (int)BS, D, FF, 1);
    k_add_ln<<<dim3((unsigned)BS), 256, 0, stream>>>(
        cur, sf, ln_g[2] + (long long)l * D, ln_b[2] + (long long)l * D, nxt, x16);
    float* tmp = cur; cur = nxt; nxt = tmp;
  }

  // ---- norm-weighted pooling + classifier head ----
  k_norm<<<dim3((unsigned)BS), 256, 0, stream>>>(cur, norms);
  k_pool<<<dim3(Bc), 256, 0, stream>>>(cur, norms, outPooled);
  k_head<<<dim3(Bc), 256, 0, stream>>>(outPooled, fc1_w, fc1_b, fc2_w, fc2_b,
                                       fc3_w, fc3_b, outLabel);
}
